// Ernie4Moe_19353122635829
// MI455X (gfx1250) — matrix operand order repaired and checked by reading, not compiled, not run on hardware
//
#include <hip/hip_runtime.h>
#include <math.h>

// Problem constants (from reference)
#define TK   2048   // tokens
#define HH   1024   // hidden
#define EE   64     // experts
#define TOPK 6
#define II   512    // expert intermediate
#define SSI  1024   // shared intermediate

typedef __bf16 bf16;
typedef __attribute__((ext_vector_type(16))) __bf16 v16bf;
typedef __attribute__((ext_vector_type(8)))  __bf16 v8bf;
typedef __attribute__((ext_vector_type(8)))  float  v8f;

// ---------------------------------------------------------------------------
// WMMA helpers (CDNA5: v_wmma_f32_16x16x32_bf16, D = A(16x32) * B(32x16) + C)
// ---------------------------------------------------------------------------
__device__ __forceinline__ v8f wmma_bf16(v16bf a, v16bf b, v8f c) {
    return __builtin_amdgcn_wmma_f32_16x16x32_bf16(
        /*neg_a=*/false, a, /*neg_b=*/false, b,
        /*c_mod=*/(short)0, c, /*reuse_a=*/false, /*reuse_b=*/false);
}

// A fragment: 16x32 bf16 tile, row-major in LDS with row stride `ld`.
// lane<16 holds M=lane, K chunks [k0..k0+7] and [k0+16..k0+23];
// lane>=16 holds M=lane-16, K chunks [k0+8..k0+15] and [k0+24..k0+31].
__device__ __forceinline__ v16bf load_a_frag(const bf16* tile, int ld, int k0, int lane) {
    int m    = lane & 15;
    int half = lane >> 4;
    const v8bf* p0 = (const v8bf*)(tile + m * ld + k0 + half * 8);
    const v8bf* p1 = (const v8bf*)(tile + m * ld + k0 + 16 + half * 8);
    v8bf lo = *p0, hi = *p1;
    v16bf a;
#pragma unroll
    for (int j = 0; j < 8; ++j) { a[j] = lo[j]; a[j + 8] = hi[j]; }
    return a;
}

// B fragment: 32x16 bf16, K-major tile in LDS (row = K, stride ldB), column
// block starting at n0. Lane l holds column N=n0+l%16; K chunks [8h..8h+7] and
// [16+8h..16+8h+7] with h = l/16 (same order as the A fragment).
__device__ __forceinline__ v16bf load_b_frag(const bf16* tile, int ldB, int n0, int lane) {
    int n    = n0 + (lane & 15);
    int half = lane >> 4;
    v16bf b;
#pragma unroll
    for (int j = 0; j < 8; ++j) {
        b[j]     = tile[(half * 8 + j) * ldB + n];
        b[j + 8] = tile[(16 + half * 8 + j) * ldB + n];
    }
    return b;
}

// Cooperative fp32 global -> bf16 LDS slice load: 32 rows x `cols` columns,
// source row stride srcLd floats. 256 threads. Optional prefetch of next slice.
__device__ __forceinline__ void load_slice_bf16(bf16* dst, const float* src, int cols,
                                                int srcLd, int tid, const float* pref) {
    if (pref) __builtin_prefetch(pref + (size_t)tid * 64, 0, 1);  // global_prefetch_b8
    int q = cols >> 2;             // float4s per row
    int total = 32 * q;
    for (int i = tid; i < total; i += 256) {
        int r = i / q;
        int c = (i - r * q) << 2;
        float4 v = *(const float4*)(src + (size_t)r * srcLd + c);
        bf16* d = dst + r * cols + c;
        d[0] = (bf16)v.x; d[1] = (bf16)v.y; d[2] = (bf16)v.z; d[3] = (bf16)v.w;
    }
}

__device__ __forceinline__ float silu_f(float g) {
    return g / (1.0f + __expf(-g));
}

// ---------------------------------------------------------------------------
// Kernel 1: fp32 -> bf16 conversion of hidden states
// ---------------------------------------------------------------------------
__global__ __launch_bounds__(256) void cvt_kernel(const float* __restrict__ x,
                                                  bf16* __restrict__ xb) {
    size_t i = ((size_t)blockIdx.x * 256 + threadIdx.x) * 4;
    float4 v = *(const float4*)(x + i);
    bf16* d = xb + i;
    d[0] = (bf16)v.x; d[1] = (bf16)v.y; d[2] = (bf16)v.z; d[3] = (bf16)v.w;
}

// ---------------------------------------------------------------------------
// Kernel 2: router — logits, sigmoid, top-6 (bias-corrected), renorm,
// bucket scatter (token id + weight) per expert.
// ---------------------------------------------------------------------------
__global__ __launch_bounds__(64) void router_kernel(const float* __restrict__ x,
                                                    const float* __restrict__ gw,
                                                    const float* __restrict__ gb,
                                                    int* __restrict__ counts,
                                                    int* __restrict__ btok,
                                                    float* __restrict__ bw) {
    int t = blockIdx.x;
    int tid = threadIdx.x;
    __shared__ __align__(16) float sx[HH];
    __shared__ float sc[EE];
    __shared__ float scorr[EE];

    for (int i = tid; i < HH / 4; i += 64)
        ((float4*)sx)[i] = ((const float4*)(x + (size_t)t * HH))[i];
    __syncthreads();

    // one expert per thread
    const float* wr = gw + (size_t)tid * HH;
    float acc = 0.f;
    for (int k = 0; k < HH; k += 4) {
        float4 w4 = *(const float4*)(wr + k);
        acc += sx[k] * w4.x + sx[k + 1] * w4.y + sx[k + 2] * w4.z + sx[k + 3] * w4.w;
    }
    float s = 1.0f / (1.0f + __expf(-acc));
    sc[tid] = s;
    scorr[tid] = s + gb[tid];
    __syncthreads();

    if (tid == 0) {
        int   idxs[TOPK];
        float wts[TOPK];
        float wsum = 0.f;
        for (int kk = 0; kk < TOPK; ++kk) {
            float best = -1e30f; int bi = 0;
            for (int e = 0; e < EE; ++e) {
                float v = scorr[e];
                if (v > best) { best = v; bi = e; }
            }
            scorr[bi] = -1e30f;
            idxs[kk] = bi;
            wts[kk] = sc[bi];
            wsum += sc[bi];
        }
        float inv = 1.0f / wsum;
        for (int kk = 0; kk < TOPK; ++kk) {
            int e = idxs[kk];
            int slot = atomicAdd(&counts[e], 1);
            btok[e * TK + slot] = t;
            bw[e * TK + slot] = wts[kk] * inv;
        }
    }
}

// ---------------------------------------------------------------------------
// Kernel 3: shared expert up/gate:  h1 = silu(x@ws_gate) * (x@ws_up)   (bf16)
// Block: 16 tokens x full SI(1024); 8 waves, each owns a 128-wide N chunk.
// ---------------------------------------------------------------------------
__global__ __launch_bounds__(256) void s1_kernel(const bf16* __restrict__ xb,
                                                 const float* __restrict__ wsg,
                                                 const float* __restrict__ wsu,
                                                 bf16* __restrict__ h1) {
    __shared__ __align__(32) bf16 lds_x[16 * HH];   // 32 KB
    __shared__ __align__(32) bf16 lds_w[32 * SSI];  // 64 KB (gate / up alternately)

    int tid  = threadIdx.x;
    int lane = tid & 31;
    int wave = tid >> 5;
    int t0   = blockIdx.x * 16;
    int nb   = wave * 128;

    for (int i = tid; i < 16 * HH / 8; i += 256)
        ((uint4*)lds_x)[i] = ((const uint4*)(xb + (size_t)t0 * HH))[i];
    __syncthreads();

    v8f accG[8] = {};
    v8f accU[8] = {};

    for (int k0 = 0; k0 < HH; k0 += 32) {
        const float* pf = (k0 + 32 < HH) ? wsg + (size_t)(k0 + 32) * SSI : nullptr;
        __syncthreads();
        load_slice_bf16(lds_w, wsg + (size_t)k0 * SSI, SSI, SSI, tid, pf);
        __syncthreads();
        v16bf af = load_a_frag(lds_x, HH, k0, lane);
#pragma unroll
        for (int s = 0; s < 8; ++s)
            accG[s] = wmma_bf16(af, load_b_frag(lds_w, SSI, nb + s * 16, lane), accG[s]);
        __syncthreads();
        load_slice_bf16(lds_w, wsu + (size_t)k0 * SSI, SSI, SSI, tid,
                        (k0 + 32 < HH) ? wsu + (size_t)(k0 + 32) * SSI : nullptr);
        __syncthreads();
#pragma unroll
        for (int s = 0; s < 8; ++s)
            accU[s] = wmma_bf16(af, load_b_frag(lds_w, SSI, nb + s * 16, lane), accU[s]);
    }

    int half = lane >> 4, nc = lane & 15;
#pragma unroll
    for (int s = 0; s < 8; ++s) {
        int n = nb + s * 16 + nc;
#pragma unroll
        for (int r = 0; r < 8; ++r) {
            int m = r + half * 8;
            float v = silu_f(accG[s][r]) * accU[s][r];
            h1[(size_t)(t0 + m) * SSI + n] = (bf16)v;
        }
    }
}

// ---------------------------------------------------------------------------
// Kernel 4: shared expert down:  out = h1 @ ws_down   (plain store, covers all)
// ---------------------------------------------------------------------------
__global__ __launch_bounds__(256) void s2_kernel(const bf16* __restrict__ h1,
                                                 const float* __restrict__ wsd,
                                                 float* __restrict__ out) {
    __shared__ __align__(32) bf16 lds_h[16 * SSI];  // 32 KB
    __shared__ __align__(32) bf16 lds_w[32 * HH];   // 64 KB

    int tid  = threadIdx.x;
    int lane = tid & 31;
    int wave = tid >> 5;
    int t0   = blockIdx.x * 16;
    int nb   = wave * 128;

    for (int i = tid; i < 16 * SSI / 8; i += 256)
        ((uint4*)lds_h)[i] = ((const uint4*)(h1 + (size_t)t0 * SSI))[i];
    __syncthreads();

    v8f accO[8] = {};
    for (int k0 = 0; k0 < SSI; k0 += 32) {
        const float* pf = (k0 + 32 < SSI) ? wsd + (size_t)(k0 + 32) * HH : nullptr;
        __syncthreads();
        load_slice_bf16(lds_w, wsd + (size_t)k0 * HH, HH, HH, tid, pf);
        __syncthreads();
        v16bf af = load_a_frag(lds_h, SSI, k0, lane);
#pragma unroll
        for (int s = 0; s < 8; ++s)
            accO[s] = wmma_bf16(af, load_b_frag(lds_w, HH, nb + s * 16, lane), accO[s]);
    }

    int half = lane >> 4, nc = lane & 15;
#pragma unroll
    for (int s = 0; s < 8; ++s) {
        int n = nb + s * 16 + nc;
#pragma unroll
        for (int r = 0; r < 8; ++r) {
            int m = r + half * 8;
            out[(size_t)(t0 + m) * HH + n] = accO[s][r];
        }
    }
}

// ---------------------------------------------------------------------------
// Kernel 5: grouped MoE expert FFN. Grid = (128 token-tiles, 64 experts).
// Phase 1: G,U[16,512] = X[16,1024] @ Wg/Wu; A = silu(G)*U*routing_w -> LDS.
// Phase 2: out += A[16,512] @ Wd[512,1024] via global atomic f32 adds.
// ---------------------------------------------------------------------------
__global__ __launch_bounds__(256) void moe_kernel(const bf16* __restrict__ xb,
                                                  const float* __restrict__ w_gate,
                                                  const float* __restrict__ w_up,
                                                  const float* __restrict__ w_down,
                                                  const int* __restrict__ counts,
                                                  const int* __restrict__ btok,
                                                  const float* __restrict__ bw,
                                                  float* __restrict__ out) {
    int e   = blockIdx.y;
    int n_e = counts[e];
    int t0  = blockIdx.x * 16;
    if (t0 >= n_e) return;

    __shared__ __align__(32) bf16 lds_x[16 * HH];  // 32 KB
    __shared__ __align__(32) bf16 lds_w[32 * HH];  // 64 KB (g+u slices / wd slice)
    __shared__ __align__(32) bf16 lds_a[16 * II];  // 16 KB
    __shared__ int   s_tok[16];
    __shared__ float s_wt[16];

    int tid  = threadIdx.x;
    int lane = tid & 31;
    int wave = tid >> 5;

    if (tid < 16) {
        int idx = t0 + tid;
        int tok = (idx < n_e) ? btok[e * TK + idx] : -1;
        s_tok[tid] = tok;
        s_wt[tid]  = (idx < n_e) ? bw[e * TK + idx] : 0.0f;
    }
    __syncthreads();

    // gather X tile (rows of gathered tokens, zero-padded)
    for (int i = tid; i < 16 * (HH / 8); i += 256) {
        int m = i >> 7;             // HH/8 = 128 uint4 per row
        int c = i & 127;
        int tok = s_tok[m];
        uint4 v = {0u, 0u, 0u, 0u};
        if (tok >= 0) v = ((const uint4*)(xb + (size_t)tok * HH))[c];
        ((uint4*)(lds_x + m * HH))[c] = v;
    }
    __syncthreads();

    // ---- phase 1: G,U (each wave owns 64 columns of I) ----
    int nb1 = wave * 64;
    const float* wgB = w_gate + (size_t)e * HH * II;
    const float* wuB = w_up   + (size_t)e * HH * II;
    bf16* lds_wg = lds_w;
    bf16* lds_wu = lds_w + 32 * II;

    v8f accG[4] = {};
    v8f accU[4] = {};
    for (int k0 = 0; k0 < HH; k0 += 32) {
        const float* pf = (k0 + 32 < HH) ? wgB + (size_t)(k0 + 32) * II : nullptr;
        __syncthreads();
        load_slice_bf16(lds_wg, wgB + (size_t)k0 * II, II, II, tid, pf);
        load_slice_bf16(lds_wu, wuB + (size_t)k0 * II, II, II, tid, nullptr);
        __syncthreads();
        v16bf af = load_a_frag(lds_x, HH, k0, lane);
#pragma unroll
        for (int s = 0; s < 4; ++s) {
            accG[s] = wmma_bf16(af, load_b_frag(lds_wg, II, nb1 + s * 16, lane), accG[s]);
            accU[s] = wmma_bf16(af, load_b_frag(lds_wu, II, nb1 + s * 16, lane), accU[s]);
        }
    }

    // A = silu(G)*U*w_tok -> lds_a
    int half = lane >> 4, nc = lane & 15;
#pragma unroll
    for (int s = 0; s < 4; ++s) {
        int n = nb1 + s * 16 + nc;
#pragma unroll
        for (int r = 0; r < 8; ++r) {
            int m = r + half * 8;
            float v = silu_f(accG[s][r]) * accU[s][r] * s_wt[m];
            lds_a[m * II + n] = (bf16)v;
        }
    }
    __syncthreads();

    // ---- phase 2: out += A @ Wd (each wave owns 128 columns of H) ----
    int nb2 = wave * 128;
    const float* wdB = w_down + (size_t)e * II * HH;
    v8f accO[8] = {};
    for (int k0 = 0; k0 < II; k0 += 32) {
        const float* pf = (k0 + 32 < II) ? wdB + (size_t)(k0 + 32) * HH : nullptr;
        __syncthreads();
        load_slice_bf16(lds_w, wdB + (size_t)k0 * HH, HH, HH, tid, pf);
        __syncthreads();
        v16bf af = load_a_frag(lds_a, II, k0, lane);
#pragma unroll
        for (int s = 0; s < 8; ++s)
            accO[s] = wmma_bf16(af, load_b_frag(lds_w, HH, nb2 + s * 16, lane), accO[s]);
    }

#pragma unroll
    for (int s = 0; s < 8; ++s) {
        int n = nb2 + s * 16 + nc;
#pragma unroll
        for (int r = 0; r < 8; ++r) {
            int m = r + half * 8;
            int tok = s_tok[m];
            if (tok >= 0)
                unsafeAtomicAdd(out + (size_t)tok * HH + n, accO[s][r]);
        }
    }
}

// ---------------------------------------------------------------------------
// Host launcher
// ---------------------------------------------------------------------------
extern "C" void kernel_launch(void* const* d_in, const int* in_sizes, int n_in,
                              void* d_out, int out_size, void* d_ws, size_t ws_size,
                              hipStream_t stream) {
    const float* x        = (const float*)d_in[0];
    const float* gate_w   = (const float*)d_in[1];
    const float* gate_b   = (const float*)d_in[2];
    const float* w_gate   = (const float*)d_in[3];
    const float* w_up     = (const float*)d_in[4];
    const float* w_down   = (const float*)d_in[5];
    const float* ws_gate  = (const float*)d_in[6];
    const float* ws_up    = (const float*)d_in[7];
    const float* ws_down  = (const float*)d_in[8];
    float* out = (float*)d_out;

    char* ws = (char*)d_ws;
    const size_t xb_bytes = (size_t)TK * HH * sizeof(bf16);   // 4 MB
    const size_t h1_bytes = (size_t)TK * SSI * sizeof(bf16);  // 4 MB
    bf16*  xb     = (bf16*)ws;
    bf16*  h1     = (bf16*)(ws + xb_bytes);
    int*   counts = (int*)(ws + xb_bytes + h1_bytes);
    int*   btok   = (int*)(ws + xb_bytes + h1_bytes + 256);
    float* bwt    = (float*)(ws + xb_bytes + h1_bytes + 256 + (size_t)EE * TK * sizeof(int));

    hipMemsetAsync(counts, 0, EE * sizeof(int), stream);

    cvt_kernel<<<(TK * HH) / (256 * 4), 256, 0, stream>>>(x, xb);
    router_kernel<<<TK, 64, 0, stream>>>(x, gate_w, gate_b, counts, btok, bwt);
    s1_kernel<<<TK / 16, 256, 0, stream>>>(xb, ws_gate, ws_up, h1);
    s2_kernel<<<TK / 16, 256, 0, stream>>>(h1, ws_down, out);
    moe_kernel<<<dim3(TK / 16, EE), 256, 0, stream>>>(xb, w_gate, w_up, w_down,
                                                      counts, btok, bwt, out);
}
